// HGATV2Layer_89034672046807
// MI455X (gfx1250) — matrix operand order repaired and checked by reading, not compiled, not run on hardware
//
#include <hip/hip_runtime.h>
#include <hip/hip_bf16.h>

typedef __attribute__((ext_vector_type(16))) _Float16 v16h;
typedef __attribute__((ext_vector_type(8)))  float    v8f;

#define DIN 128

// ---------------- helpers ----------------
__device__ __forceinline__ unsigned ord_encode(float f) {
  unsigned u = __float_as_uint(f);
  return (u & 0x80000000u) ? ~u : (u | 0x80000000u);
}
__device__ __forceinline__ float ord_decode(unsigned k) {
  unsigned u = (k & 0x80000000u) ? (k & 0x7FFFFFFFu) : ~k;
  return __uint_as_float(u);
}
__device__ __forceinline__ float lrelu(float x) { return x > 0.f ? x : 0.2f * x; }

// CDNA5 async Global->LDS copy (ASYNCcnt-tracked, no VGPR data movement).
// vdst = 32-bit LDS byte address, vaddr = 64-bit global address.
__device__ __forceinline__ void async_copy_b128(unsigned lds_addr, const void* gaddr) {
  asm volatile("global_load_async_to_lds_b128 %0, %1, off"
               :: "v"(lds_addr), "v"(gaddr)
               : "memory");
}
__device__ __forceinline__ void wait_async0() {
  asm volatile("s_wait_asynccnt 0" ::: "memory");
}

__global__ void fill_u32_kernel(unsigned* __restrict__ p, unsigned v, long long n) {
  long long i = (long long)blockIdx.x * blockDim.x + threadIdx.x;
  if (i < n) p[i] = v;
}

// ---------------- weight packing into WMMA B-fragment layout ----------------
// P[((nt*4+ks)*32 + lane)*16 + h] = f16( W[K][nt*16 + lane%16] ),
//   K = ks*32 + (lane<16 ? 0 : 8) + (h<8 ? h : h+8)
__global__ void pack_w_kernel(const float* __restrict__ W, _Float16* __restrict__ P, int ncols) {
  int idx = blockIdx.x * blockDim.x + threadIdx.x;
  int total = ncols * DIN;
  if (idx >= total) return;
  int h    = idx & 15;
  int lane = (idx >> 4) & 31;
  int frag = idx >> 9;          // nt*4 + ks
  int ks   = frag & 3;
  int nt   = frag >> 2;
  int col  = nt * 16 + (lane & 15);
  int K    = ks * 32 + ((lane & 16) ? 8 : 0) + (h < 8 ? h : h + 8);
  P[idx] = (_Float16)W[K * ncols + col];
}

// ---------------- WMMA GEMM: Y[nrows,NCOLS] = X[nrows,128] @ W ----------------
// block = 256 threads = 8 waves; each wave computes a 16-row slab across all N tiles.
template <int NCOLS>
__global__ __launch_bounds__(256) void gemm_wmma_kernel(const float* __restrict__ X,
                                                        const _Float16* __restrict__ P,
                                                        float* __restrict__ Y, int nrows) {
  __shared__ _Float16 Bs[NCOLS * DIN] __attribute__((aligned(32)));
  {   // stage packed B into LDS via async Global->LDS b128 copies
    const unsigned lds_base = (unsigned)(size_t)(void*)Bs;   // flat->LDS: addr[31:0]
    const char* g = (const char*)P;
    const int chunks = NCOLS * DIN * 2 / 16;                 // 16B chunks, multiple of 256
    for (int i = threadIdx.x; i < chunks; i += blockDim.x)
      async_copy_b128(lds_base + (unsigned)i * 16u, g + (size_t)i * 16u);
    wait_async0();
  }
  __syncthreads();

  const int lane = threadIdx.x & 31;
  const int wave = threadIdx.x >> 5;
  const int row0 = blockIdx.x * 128 + wave * 16;
  int lrow = row0 + (lane & 15);
  if (lrow >= nrows) lrow = nrows - 1;          // clamp; stores are predicated
  const float* xrow = X + (size_t)lrow * DIN;
  const int hi = (lane & 16) ? 8 : 0;

  // A fragments for all 4 k-steps (16-bit A 16x32 layout):
  // lanes 0-15 row M=lane, halves h<8 -> K=ks*32+h, h>=8 -> K=ks*32+8+h
  // lanes16-31 row M=lane-16, shifted by +8
  v16h a[4];
#pragma unroll
  for (int ks = 0; ks < 4; ++ks) {
    const int K0 = ks * 32 + hi;
    float4 f0 = *(const float4*)(xrow + K0);
    float4 f1 = *(const float4*)(xrow + K0 + 4);
    float4 f2 = *(const float4*)(xrow + K0 + 16);
    float4 f3 = *(const float4*)(xrow + K0 + 20);
    float t[16] = {f0.x, f0.y, f0.z, f0.w, f1.x, f1.y, f1.z, f1.w,
                   f2.x, f2.y, f2.z, f2.w, f3.x, f3.y, f3.z, f3.w};
    v16h av;
#pragma unroll
    for (int i = 0; i < 16; ++i) av[i] = (_Float16)t[i];
    a[ks] = av;
  }

  for (int nt = 0; nt < NCOLS / 16; ++nt) {
    // preload all 4 B fragments so the ds_loads pipeline and WMMAs chain
    const _Float16* bbase = Bs + (size_t)nt * 4 * 32 * 16 + (size_t)lane * 16;
    v16h b0 = *(const v16h*)(bbase + 0 * 512);
    v16h b1 = *(const v16h*)(bbase + 1 * 512);
    v16h b2 = *(const v16h*)(bbase + 2 * 512);
    v16h b3 = *(const v16h*)(bbase + 3 * 512);
    v8f acc = {};
    acc = __builtin_amdgcn_wmma_f32_16x16x32_f16(false, a[0], false, b0, (short)0, acc, false, false);
    acc = __builtin_amdgcn_wmma_f32_16x16x32_f16(false, a[1], false, b1, (short)0, acc, false, false);
    acc = __builtin_amdgcn_wmma_f32_16x16x32_f16(false, a[2], false, b2, (short)0, acc, false, false);
    acc = __builtin_amdgcn_wmma_f32_16x16x32_f16(false, a[3], false, b3, (short)0, acc, false, false);

    const int col = nt * 16 + (lane & 15);
    float* yp = Y + (size_t)(row0 + hi) * NCOLS + col;
    if (row0 + 16 <= nrows) {        // wave-uniform fast path: no predication
#pragma unroll
      for (int r = 0; r < 8; ++r) yp[(size_t)r * NCOLS] = acc[r];
    } else {                          // tail block only
#pragma unroll
      for (int r = 0; r < 8; ++r)
        if (row0 + hi + r < nrows) yp[(size_t)r * NCOLS] = acc[r];
    }
  }
}

// ---------------- edge pass 1: logits + segment max ----------------
__global__ void edge_score_max_kernel(const float* __restrict__ xl, const float* __restrict__ xr,
                                      const int* __restrict__ src, const int* __restrict__ dst,
                                      const float* __restrict__ att,
                                      float* __restrict__ pbuf, unsigned* __restrict__ smax,
                                      int nE, int nLoops) {
  int idx = blockIdx.x * blockDim.x + threadIdx.x;
  int total = (nE + nLoops) * 4;
  if (idx >= total) return;
  int h = idx & 3, e = idx >> 2;
  int s, d;
  if (e < nE) { s = src[e]; d = dst[e]; } else { s = d = e - nE; }
  const float4* a = (const float4*)(xl + (size_t)s * 256 + h * 64);
  const float4* b = (const float4*)(xr + (size_t)d * 256 + h * 64);
  const float4* w = (const float4*)(att + h * 64);
  float acc = 0.f;
#pragma unroll 4
  for (int i = 0; i < 16; ++i) {
    float4 av = a[i], bv = b[i], wv = w[i];
    acc += lrelu(av.x + bv.x) * wv.x;
    acc += lrelu(av.y + bv.y) * wv.y;
    acc += lrelu(av.z + bv.z) * wv.z;
    acc += lrelu(av.w + bv.w) * wv.w;
  }
  pbuf[idx] = acc;
  atomicMax(&smax[d * 4 + h], ord_encode(acc));
}

// ---------------- edge pass 2: p = exp(s - smax); denom += p ----------------
__global__ void edge_exp_kernel(float* __restrict__ pbuf, const unsigned* __restrict__ smax,
                                float* __restrict__ denom, const int* __restrict__ dst,
                                int nE, int nLoops) {
  int idx = blockIdx.x * blockDim.x + threadIdx.x;
  int total = (nE + nLoops) * 4;
  if (idx >= total) return;
  int h = idx & 3, e = idx >> 2;
  int d = (e < nE) ? dst[e] : e - nE;
  float mx = ord_decode(smax[d * 4 + h]);
  if (!isfinite(mx)) mx = 0.f;
  float p = expf(pbuf[idx] - mx);
  pbuf[idx] = p;
  atomicAdd(&denom[d * 4 + h], p);
}

// ---------------- edge pass 3: acc[dst] += p * xl[src] ----------------
__global__ void edge_scatter_kernel(const float* __restrict__ pbuf, const float* __restrict__ xl,
                                    const int* __restrict__ src, const int* __restrict__ dst,
                                    float* __restrict__ acc, int nE, int nLoops) {
  int idx = blockIdx.x * blockDim.x + threadIdx.x;
  int total = (nE + nLoops) * 64;
  if (idx >= total) return;
  int j = idx & 63, e = idx >> 6;
  int s, d;
  if (e < nE) { s = src[e]; d = dst[e]; } else { s = d = e - nE; }
  int h = j >> 4;
  float p = pbuf[e * 4 + h];
  float4 v = *(const float4*)(xl + (size_t)s * 256 + j * 4);
  float* ap = acc + (size_t)d * 256 + j * 4;
  atomicAdd(ap + 0, p * v.x);
  atomicAdd(ap + 1, p * v.y);
  atomicAdd(ap + 2, p * v.z);
  atomicAdd(ap + 3, p * v.w);
}

// ---------------- finalize: head mean / denom + residual + bias ----------------
__global__ void finalize_kernel(const float* __restrict__ acc, const float* __restrict__ denom,
                                const float* __restrict__ res, const float* __restrict__ bias,
                                float* __restrict__ out, int N) {
  int idx = blockIdx.x * blockDim.x + threadIdx.x;
  if (idx >= N * 64) return;
  int c = idx & 63, n = idx >> 6;
  float v = 0.f;
#pragma unroll
  for (int h = 0; h < 4; ++h)
    v += acc[(size_t)n * 256 + h * 64 + c] / fmaxf(denom[n * 4 + h], 1e-16f);
  out[idx] = 0.25f * v + res[(size_t)n * 64 + c] + bias[c];
}

// ---------------- batch-norm: stats (atomic) then apply+ELU ----------------
__global__ void bn_stats_kernel(const float* __restrict__ A, const float* __restrict__ B,
                                float* __restrict__ stats, int N) {
  int gid = blockIdx.x * blockDim.x + threadIdx.x;   // 64*1024 threads
  int c = gid & 63;
  float s = 0.f, s2 = 0.f;
  for (int n = gid >> 6; n < N; n += 1024) {
    float v = A[(size_t)n * 64 + c];
    if (B) v = fmaxf(v, B[(size_t)n * 64 + c]);
    s += v; s2 += v * v;
  }
  atomicAdd(&stats[c], s);
  atomicAdd(&stats[64 + c], s2);
}

__global__ void bn_apply_kernel(const float* __restrict__ A, const float* __restrict__ B,
                                const float* __restrict__ stats,
                                const float* __restrict__ gamma, const float* __restrict__ beta,
                                float* __restrict__ out, int N) {
  int idx = blockIdx.x * blockDim.x + threadIdx.x;
  if (idx >= N * 64) return;
  int c = idx & 63;
  float v = A[idx];
  if (B) v = fmaxf(v, B[idx]);
  float mu  = stats[c] / (float)N;
  float var = stats[64 + c] / (float)N - mu * mu;
  float xn = (v - mu) * rsqrtf(var + 1e-5f) * gamma[c] + beta[c];
  out[idx] = xn > 0.f ? xn : expf(xn) - 1.f;   // ELU(alpha=1)
}

// ---------------- orchestration ----------------
extern "C" void kernel_launch(void* const* d_in, const int* in_sizes, int n_in,
                              void* d_out, int out_size, void* d_ws, size_t ws_size,
                              hipStream_t stream) {
  const float* x_op = (const float*)d_in[0];
  const float* x_m  = (const float*)d_in[1];
  const int*  e_oo  = (const int*)d_in[2];     // [2,E_oo]
  const int* src_om = (const int*)d_in[3];
  const int* dst_om = (const int*)d_in[4];
  const int* src_mo = (const int*)d_in[5];
  const int* dst_mo = (const int*)d_in[6];
  const float* Wl[3]   = {(const float*)d_in[7],  (const float*)d_in[12], (const float*)d_in[17]};
  const float* Wr[3]   = {(const float*)d_in[8],  (const float*)d_in[13], (const float*)d_in[18]};
  const float* att[3]  = {(const float*)d_in[9],  (const float*)d_in[14], (const float*)d_in[19]};
  const float* bias[3] = {(const float*)d_in[10], (const float*)d_in[15], (const float*)d_in[20]};
  const float* Wres[3] = {(const float*)d_in[11], (const float*)d_in[16], (const float*)d_in[21]};
  const float* bn_g_op = (const float*)d_in[22];
  const float* bn_b_op = (const float*)d_in[23];
  const float* bn_g_m  = (const float*)d_in[24];
  const float* bn_b_m  = (const float*)d_in[25];

  const int n_op = in_sizes[0] / DIN;
  const int n_m  = in_sizes[1] / DIN;
  const int E_oo = in_sizes[2] / 2;
  const int E_om = in_sizes[3];
  const int E_mo = in_sizes[5];
  int maxE = E_oo + n_op;
  if (E_om > maxE) maxE = E_om;
  if (E_mo > maxE) maxE = E_mo;

  // ---- carve workspace ----
  char* wsb = (char*)d_ws;
  size_t off = 0;
  auto carve = [&](size_t bytes) -> void* {
    off = (off + 255) & ~(size_t)255;
    void* p = wsb + off;
    off += bytes;
    return p;
  };
  _Float16* pWl[3]; _Float16* pWr[3]; _Float16* pWres[3];
  for (int t = 0; t < 3; ++t) {
    pWl[t]   = (_Float16*)carve((size_t)DIN * 256 * 2);
    pWr[t]   = (_Float16*)carve((size_t)DIN * 256 * 2);
    pWres[t] = (_Float16*)carve((size_t)DIN * 64 * 2);
  }
  float*    xlbuf   = (float*)carve((size_t)n_op * 256 * 4);
  float*    xrbuf   = (float*)carve((size_t)n_op * 256 * 4);
  float*    accbuf  = (float*)carve((size_t)n_op * 256 * 4);
  float*    resbuf  = (float*)carve((size_t)n_op * 64 * 4);
  float*    pbuf    = (float*)carve((size_t)maxE * 4 * 4);
  unsigned* smaxbuf = (unsigned*)carve((size_t)n_op * 4 * 4);
  float*    denombuf= (float*)carve((size_t)n_op * 4 * 4);
  float*    st_oo   = (float*)carve((size_t)n_op * 64 * 4);
  float*    st_mo   = (float*)carve((size_t)n_op * 64 * 4);
  float*    st_om   = (float*)carve((size_t)n_m  * 64 * 4);
  float*    stats   = (float*)carve(128 * 4);

  // ---- pack weights to f16 WMMA fragment layout ----
  for (int t = 0; t < 3; ++t) {
    pack_w_kernel<<<(256 * DIN + 255) / 256, 256, 0, stream>>>(Wl[t],   pWl[t],   256);
    pack_w_kernel<<<(256 * DIN + 255) / 256, 256, 0, stream>>>(Wr[t],   pWr[t],   256);
    pack_w_kernel<<<(64  * DIN + 255) / 256, 256, 0, stream>>>(Wres[t], pWres[t], 64);
  }

  auto run_conv = [&](const float* xs, int Ns, const float* xd, int Nd,
                      const int* src, const int* dst, int nE, int nLoops,
                      const _Float16* wl, const _Float16* wr, const _Float16* wres,
                      const float* at, const float* bi, float* outStage) {
    gemm_wmma_kernel<256><<<(Ns + 127) / 128, 256, 0, stream>>>(xs, wl, xlbuf, Ns);
    gemm_wmma_kernel<256><<<(Nd + 127) / 128, 256, 0, stream>>>(xd, wr, xrbuf, Nd);
    gemm_wmma_kernel<64><<<(Nd + 127) / 128, 256, 0, stream>>>(xd, wres, resbuf, Nd);
    long long nsm = (long long)Nd * 4;
    fill_u32_kernel<<<(unsigned)((nsm + 255) / 256), 256, 0, stream>>>(smaxbuf, 0x007FFFFFu, nsm); // enc(-inf)
    fill_u32_kernel<<<(unsigned)((nsm + 255) / 256), 256, 0, stream>>>((unsigned*)denombuf, 0u, nsm);
    long long nacc = (long long)Nd * 256;
    fill_u32_kernel<<<(unsigned)((nacc + 255) / 256), 256, 0, stream>>>((unsigned*)accbuf, 0u, nacc);
    long long tE = (long long)(nE + nLoops);
    edge_score_max_kernel<<<(unsigned)((tE * 4 + 255) / 256), 256, 0, stream>>>(
        xlbuf, xrbuf, src, dst, at, pbuf, smaxbuf, nE, nLoops);
    edge_exp_kernel<<<(unsigned)((tE * 4 + 255) / 256), 256, 0, stream>>>(
        pbuf, smaxbuf, denombuf, dst, nE, nLoops);
    edge_scatter_kernel<<<(unsigned)((tE * 64 + 255) / 256), 256, 0, stream>>>(
        pbuf, xlbuf, src, dst, accbuf, nE, nLoops);
    finalize_kernel<<<(unsigned)(((long long)Nd * 64 + 255) / 256), 256, 0, stream>>>(
        accbuf, denombuf, resbuf, bi, outStage, Nd);
  };

  // (op,to,op) with self loops
  run_conv(x_op, n_op, x_op, n_op, e_oo, e_oo + E_oo, E_oo, n_op,
           pWl[0], pWr[0], pWres[0], att[0], bias[0], st_oo);
  // (machine,rev_on,op)
  run_conv(x_m, n_m, x_op, n_op, src_mo, dst_mo, E_mo, 0,
           pWl[2], pWr[2], pWres[2], att[2], bias[2], st_mo);
  // (op,on,machine)
  run_conv(x_op, n_op, x_m, n_m, src_om, dst_om, E_om, 0,
           pWl[1], pWr[1], pWres[1], att[1], bias[1], st_om);

  float* out_op = (float*)d_out;
  float* out_m  = (float*)d_out + (size_t)n_op * 64;

  // op destination: max(st_oo, st_mo) -> BN -> ELU
  fill_u32_kernel<<<1, 256, 0, stream>>>((unsigned*)stats, 0u, 128);
  bn_stats_kernel<<<256, 256, 0, stream>>>(st_oo, st_mo, stats, n_op);
  bn_apply_kernel<<<(unsigned)(((long long)n_op * 64 + 255) / 256), 256, 0, stream>>>(
      st_oo, st_mo, stats, bn_g_op, bn_b_op, out_op, n_op);

  // machine destination
  fill_u32_kernel<<<1, 256, 0, stream>>>((unsigned*)stats, 0u, 128);
  bn_stats_kernel<<<256, 256, 0, stream>>>(st_om, nullptr, stats, n_m);
  bn_apply_kernel<<<(unsigned)(((long long)n_m * 64 + 255) / 256), 256, 0, stream>>>(
      st_om, nullptr, stats, bn_g_m, bn_b_m, out_m, n_m);
}
